// _Encoder_54485955117829
// MI455X (gfx1250) — matrix operand order repaired and checked by reading, not compiled, not run on hardware
//
#include <hip/hip_runtime.h>
#include <hip/hip_bf16.h>

// ---------------- problem constants ----------------
#define T_ 512
#define B_ 64
#define M_ 512
#define DM_ 384
#define DX_ 384
#define H_ 192
#define A_ 192
#define XC_ (DX_ + DM_)   // 768
#define G3_ (3 * H_)      // 576

typedef __attribute__((ext_vector_type(16))) _Float16 v16h;
typedef __attribute__((ext_vector_type(8)))  _Float16 v8h;
typedef __attribute__((ext_vector_type(8)))  float    v8f;

// ---------------------------------------------------------------------------
// Generic WMMA f16 GEMM:  C[Mrows x N] = A[Mrows x K] * Bt[N x K]^T (+ bias)
// A row-major f16, Bt = B pre-transposed (row n holds K contiguous values).
// One wave (32 threads) computes one 16x16 tile; K must be a multiple of 32.
// Optional f32 and/or f16 outputs.
// ---------------------------------------------------------------------------
__global__ void gemm16_wmma(const _Float16* __restrict__ A,
                            const _Float16* __restrict__ Bt,
                            float* __restrict__ Cf,
                            _Float16* __restrict__ Ch,
                            const float* __restrict__ bias,
                            int N, int K)
{
    const int n0   = blockIdx.x * 16;
    const int m0   = blockIdx.y * 16;
    const int lane = threadIdx.x;      // 0..31, wave32
    const int r    = lane & 15;
    const int hi   = lane >> 4;

    const _Float16* arow = A  + (size_t)(m0 + r) * K;
    const _Float16* bcol = Bt + (size_t)(n0 + r) * K;

    v8f acc = {};
    for (int k0 = 0; k0 < K; k0 += 32) {
        // A fragment (16-bit A-matrix 16x32 layout):
        //   hi=0: V0..3 = K[k0+0..7], V4..7 = K[k0+16..23]
        //   hi=1: V0..3 = K[k0+8..15], V4..7 = K[k0+24..31]
        v8h a0 = *(const v8h*)(arow + k0 + hi * 8);
        v8h a1 = *(const v8h*)(arow + k0 + 16 + hi * 8);
        // B fragment (32x16): same K order as A: hi=0: K[k0+0..7],K[k0+16..23]; hi=1: K[k0+8..15],K[k0+24..31],
        // column = lane%16; transposed storage makes this contiguous.
        v8h b0 = *(const v8h*)(bcol + k0 + hi * 8);
        v8h b1 = *(const v8h*)(bcol + k0 + 16 + hi * 8);
        v16h av, bv;
#pragma unroll
        for (int i = 0; i < 8; ++i) {
            av[i] = a0[i]; av[i + 8] = a1[i];
            bv[i] = b0[i]; bv[i + 8] = b1[i];
        }
        acc = __builtin_amdgcn_wmma_f32_16x16x32_f16(
            /*neg_a=*/false, av, /*neg_b=*/false, bv,
            /*c_mod=*/(short)0, acc, /*reuse_a=*/false, /*reuse_b=*/false);
    }

    const float bv0 = bias ? bias[n0 + r] : 0.0f;
#pragma unroll
    for (int i = 0; i < 8; ++i) {
        const int m = m0 + hi * 8 + i;            // C/D layout: VGPR i -> row i (+8 for hi)
        const size_t idx = (size_t)m * N + n0 + r;
        const float val = acc[i] + bv0;
        if (Cf) Cf[idx] = val;
        if (Ch) Ch[idx] = (_Float16)val;
    }
}

// ---------------- conversion helpers ----------------
__global__ void cvt_f32_f16(const float* __restrict__ in, _Float16* __restrict__ out, int n)
{
    int i = blockIdx.x * blockDim.x + threadIdx.x;
    if (i < n) out[i] = (_Float16)in[i];
}

// in: [K][N] f32 row-major -> out: [N][K] f16 (transposed for B-fragment loads)
__global__ void cvt_tr_f16(const float* __restrict__ in, _Float16* __restrict__ out, int K, int N)
{
    int n = blockIdx.x * 16 + threadIdx.x;
    int k = blockIdx.y * 16 + threadIdx.y;
    if (n < N && k < K) out[(size_t)n * K + k] = (_Float16)in[(size_t)k * N + n];
}

// ---------------------------------------------------------------------------
// Attention scores: one wave per (m,b).
// s[m*B+b] = sum_a v[a] * tanh(memP[m,b,a] + xP[t,b,a] + hW[b,a]), masked.
// ---------------------------------------------------------------------------
__global__ void attn_scores(const _Float16* __restrict__ memP,
                            const _Float16* __restrict__ xPt,   // already offset to step t
                            const float* __restrict__ hW,
                            const float* __restrict__ v,
                            const unsigned char* __restrict__ mask,
                            float* __restrict__ s)
{
    const int wid  = (int)((blockIdx.x * blockDim.x + threadIdx.x) >> 5); // (m,b) index
    const int lane = threadIdx.x & 31;
    const int b    = wid & (B_ - 1);

    const _Float16* mp = memP + (size_t)wid * A_;
    const _Float16* xp = xPt  + (size_t)b   * A_;
    const float*    hw = hW   + (size_t)b   * A_;

    float sum = 0.0f;
#pragma unroll
    for (int a = lane; a < A_; a += 32)
        sum += v[a] * tanhf((float)mp[a] + (float)xp[a] + hw[a]);

#pragma unroll
    for (int off = 16; off > 0; off >>= 1)
        sum += __shfl_xor(sum, off, 32);

    if (lane == 0) {
        if (!mask[wid]) sum = -1e9f;
        s[wid] = sum;
    }
}

// ---------------------------------------------------------------------------
// Fused softmax over M (per batch column) + context + concat into xc16.
// xc16[b][0:DX] = x16[t,b,:] ;  xc16[b][DX:DX+DM] = sum_m a[m]*memory[m,b,:]
// ---------------------------------------------------------------------------
__global__ void softmax_ctx(const float* __restrict__ s,
                            const float* __restrict__ memory,
                            const _Float16* __restrict__ x16t,  // offset to step t
                            _Float16* __restrict__ xc16)
{
    const int b   = blockIdx.x;
    const int tid = threadIdx.x;          // 256 threads
    __shared__ float sa[M_];
    __shared__ float red[256];

    float mx = -1e30f;
    for (int m = tid; m < M_; m += 256) {
        float val = s[m * B_ + b];
        sa[m] = val;
        mx = fmaxf(mx, val);
    }
    red[tid] = mx; __syncthreads();
    for (int o = 128; o > 0; o >>= 1) {
        if (tid < o) red[tid] = fmaxf(red[tid], red[tid + o]);
        __syncthreads();
    }
    mx = red[0]; __syncthreads();

    float ss = 0.0f;
    for (int m = tid; m < M_; m += 256) {
        float e = __expf(sa[m] - mx);
        sa[m] = e;
        ss += e;
    }
    red[tid] = ss; __syncthreads();
    for (int o = 128; o > 0; o >>= 1) {
        if (tid < o) red[tid] += red[tid + o];
        __syncthreads();
    }
    const float inv = 1.0f / red[0];
    __syncthreads();
    for (int m = tid; m < M_; m += 256) sa[m] *= inv;
    __syncthreads();

    // concat: input slice
    for (int j = tid; j < DX_; j += 256)
        xc16[(size_t)b * XC_ + j] = x16t[(size_t)b * DX_ + j];

    // context: coalesced over d, L2-resident stream over m
    for (int d = tid; d < DM_; d += 256) {
        const float* mem_bd = memory + (size_t)b * DM_ + d;
        float acc = 0.0f;
#pragma unroll 4
        for (int m = 0; m < M_; ++m)
            acc += sa[m] * mem_bd[(size_t)m * B_ * DM_];
        xc16[(size_t)b * XC_ + DX_ + d] = (_Float16)acc;
    }
}

// xcg = sigmoid(g) * xc   (R-net input gate), f16 out for next WMMA GEMM
__global__ void gate_apply(const float* __restrict__ g,
                           const _Float16* __restrict__ xc16,
                           _Float16* __restrict__ xcg16)
{
    int i = blockIdx.x * blockDim.x + threadIdx.x;    // B_*XC_
    float sg = 1.0f / (1.0f + __expf(-g[i]));
    xcg16[i] = (_Float16)(sg * (float)xc16[i]);
}

// GRU cell update (biases already fused into gi/gh by the GEMM epilogue)
__global__ void gru_update(const float* __restrict__ gi,
                           const float* __restrict__ gh,
                           float* __restrict__ h,
                           _Float16* __restrict__ h16,
                           float* __restrict__ outt)   // offset to out[t][0][dir*H]
{
    int idx = blockIdx.x * blockDim.x + threadIdx.x;   // B_*H_
    int b = idx / H_, j = idx - b * H_;
    const float* gib = gi + (size_t)b * G3_;
    const float* ghb = gh + (size_t)b * G3_;
    float r = 1.0f / (1.0f + __expf(-(gib[j] + ghb[j])));
    float z = 1.0f / (1.0f + __expf(-(gib[H_ + j] + ghb[H_ + j])));
    float n = tanhf(gib[2 * H_ + j] + r * ghb[2 * H_ + j]);
    float hp = h[idx];
    float hn = (1.0f - z) * n + z * hp;
    h[idx] = hn;
    h16[idx] = (_Float16)hn;
    outt[(size_t)b * (2 * H_) + j] = hn;
}

// ---------------------------------------------------------------------------
extern "C" void kernel_launch(void* const* d_in, const int* in_sizes, int n_in,
                              void* d_out, int out_size, void* d_ws, size_t ws_size,
                              hipStream_t stream)
{
    (void)in_sizes; (void)n_in; (void)out_size; (void)ws_size;

    const float*         memory = (const float*)d_in[0];
    const unsigned char* mask   = (const unsigned char*)d_in[1];
    const float*         input  = (const float*)d_in[2];
    struct DirW { const float *Wm, *Wx, *Wh, *v, *Wg, *Wih, *Whh, *bih, *bhh; };
    DirW wd[2];
    for (int d = 0; d < 2; ++d) {
        int o = 3 + d * 9;
        wd[d].Wm  = (const float*)d_in[o + 0];
        wd[d].Wx  = (const float*)d_in[o + 1];
        wd[d].Wh  = (const float*)d_in[o + 2];
        wd[d].v   = (const float*)d_in[o + 3];
        wd[d].Wg  = (const float*)d_in[o + 4];
        wd[d].Wih = (const float*)d_in[o + 5];
        wd[d].Whh = (const float*)d_in[o + 6];
        wd[d].bih = (const float*)d_in[o + 7];
        wd[d].bhh = (const float*)d_in[o + 8];
    }
    float* out = (float*)d_out;

    // ---- workspace carve-out (deterministic pointer math) ----
    char* wbase = (char*)d_ws;
    size_t off = 0;
    auto alloc = [&](size_t bytes) -> void* {
        void* p = wbase + off;
        off += (bytes + 255) & ~(size_t)255;
        return p;
    };
    _Float16* x16    = (_Float16*)alloc((size_t)T_ * B_ * DX_ * 2);
    _Float16* mem16  = (_Float16*)alloc((size_t)M_ * B_ * DM_ * 2);
    _Float16* memP16[2], *xP16[2];
    _Float16* Wm16t[2], *Wx16t[2], *Wh16t[2], *Wg16t[2], *Wih16t[2], *Whh16t[2];
    float *hW[2], *s[2], *g[2], *gi[2], *gh[2], *h[2];
    _Float16 *xc16[2], *xcg16[2], *h16[2];
    for (int d = 0; d < 2; ++d) {
        memP16[d] = (_Float16*)alloc((size_t)M_ * B_ * A_ * 2);
        xP16[d]   = (_Float16*)alloc((size_t)T_ * B_ * A_ * 2);
        Wm16t[d]  = (_Float16*)alloc((size_t)A_ * DM_ * 2);
        Wx16t[d]  = (_Float16*)alloc((size_t)A_ * DX_ * 2);
        Wh16t[d]  = (_Float16*)alloc((size_t)A_ * H_ * 2);
        Wg16t[d]  = (_Float16*)alloc((size_t)XC_ * XC_ * 2);
        Wih16t[d] = (_Float16*)alloc((size_t)G3_ * XC_ * 2);
        Whh16t[d] = (_Float16*)alloc((size_t)G3_ * H_ * 2);
        hW[d]    = (float*)alloc((size_t)B_ * A_ * 4);
        s[d]     = (float*)alloc((size_t)M_ * B_ * 4);
        g[d]     = (float*)alloc((size_t)B_ * XC_ * 4);
        gi[d]    = (float*)alloc((size_t)B_ * G3_ * 4);
        gh[d]    = (float*)alloc((size_t)B_ * G3_ * 4);
        h[d]     = (float*)alloc((size_t)B_ * H_ * 4);
        xc16[d]  = (_Float16*)alloc((size_t)B_ * XC_ * 2);
        xcg16[d] = (_Float16*)alloc((size_t)B_ * XC_ * 2);
        h16[d]   = (_Float16*)alloc((size_t)B_ * H_ * 2);
    }

    // ---- one-time conversions ----
    {
        int n = T_ * B_ * DX_;
        cvt_f32_f16<<<(n + 255) / 256, 256, 0, stream>>>(input, x16, n);
        n = M_ * B_ * DM_;
        cvt_f32_f16<<<(n + 255) / 256, 256, 0, stream>>>(memory, mem16, n);
    }
    dim3 tb(16, 16);
    for (int d = 0; d < 2; ++d) {
        cvt_tr_f16<<<dim3(A_ / 16, DM_ / 16), tb, 0, stream>>>(wd[d].Wm,  Wm16t[d],  DM_, A_);
        cvt_tr_f16<<<dim3(A_ / 16, DX_ / 16), tb, 0, stream>>>(wd[d].Wx,  Wx16t[d],  DX_, A_);
        cvt_tr_f16<<<dim3(A_ / 16, H_  / 16), tb, 0, stream>>>(wd[d].Wh,  Wh16t[d],  H_,  A_);
        cvt_tr_f16<<<dim3(XC_ / 16, XC_ / 16), tb, 0, stream>>>(wd[d].Wg,  Wg16t[d],  XC_, XC_);
        cvt_tr_f16<<<dim3(G3_ / 16, XC_ / 16), tb, 0, stream>>>(wd[d].Wih, Wih16t[d], XC_, G3_);
        cvt_tr_f16<<<dim3(G3_ / 16, H_  / 16), tb, 0, stream>>>(wd[d].Whh, Whh16t[d], H_,  G3_);
        // hoisted projections (WMMA): memP = mem @ Wm, xP = x @ Wx  (f16 outputs)
        gemm16_wmma<<<dim3(A_ / 16, (M_ * B_) / 16), 32, 0, stream>>>(
            mem16, Wm16t[d], nullptr, memP16[d], nullptr, A_, DM_);
        gemm16_wmma<<<dim3(A_ / 16, (T_ * B_) / 16), 32, 0, stream>>>(
            x16, Wx16t[d], nullptr, xP16[d], nullptr, A_, DX_);
        hipMemsetAsync(h[d],   0, (size_t)B_ * H_ * 4, stream);
        hipMemsetAsync(h16[d], 0, (size_t)B_ * H_ * 2, stream);
    }

    // ---- sequential recurrence: 512 steps x 2 directions ----
    for (int t = 0; t < T_; ++t) {
        for (int d = 0; d < 2; ++d) {
            const int tx = d ? (T_ - 1 - t) : t;   // bw step k consumes/produces index T-1-k
            // hW = h @ Wh
            gemm16_wmma<<<dim3(A_ / 16, B_ / 16), 32, 0, stream>>>(
                h16[d], Wh16t[d], hW[d], nullptr, nullptr, A_, H_);
            // attention scores: wave per (m,b)
            attn_scores<<<(M_ * B_) / 8, 256, 0, stream>>>(
                memP16[d], xP16[d] + (size_t)tx * B_ * A_, hW[d], wd[d].v, mask, s[d]);
            // softmax over M + context + concat
            softmax_ctx<<<B_, 256, 0, stream>>>(
                s[d], memory, x16 + (size_t)tx * B_ * DX_, xc16[d]);
            // input gate: g = xc @ Wg ; xcg = sigmoid(g)*xc
            gemm16_wmma<<<dim3(XC_ / 16, B_ / 16), 32, 0, stream>>>(
                xc16[d], Wg16t[d], g[d], nullptr, nullptr, XC_, XC_);
            gate_apply<<<(B_ * XC_) / 256, 256, 0, stream>>>(g[d], xc16[d], xcg16[d]);
            // gi = xcg @ Wih + bih ; gh = h @ Whh + bhh  (bias fused in epilogue)
            gemm16_wmma<<<dim3(G3_ / 16, B_ / 16), 32, 0, stream>>>(
                xcg16[d], Wih16t[d], gi[d], nullptr, wd[d].bih, G3_, XC_);
            gemm16_wmma<<<dim3(G3_ / 16, B_ / 16), 32, 0, stream>>>(
                h16[d], Whh16t[d], gh[d], nullptr, wd[d].bhh, G3_, H_);
            // GRU update + write output slice
            gru_update<<<(B_ * H_) / 256, 256, 0, stream>>>(
                gi[d], gh[d], h[d], h16[d],
                out + (size_t)tx * B_ * 2 * H_ + (size_t)d * H_);
        }
    }
}
